// SupConLoss_6408091205873
// MI455X (gfx1250) — matrix operand order repaired and checked by reading, not compiled, not run on hardware
//
#include <hip/hip_runtime.h>
#include <hip/hip_bf16.h>
#include <math.h>

// ---------------- problem constants ----------------
#define BB 32
#define SS 196
#define DD 256
#define KK 100
#define NN (BB * SS)          // 6272
#define NTILES (NN / 16)      // 392 (exact, = 8 waves * 49)
#define KSTEPS (DD / 32)      // 8 k-steps of 32 for bf16 WMMA
#define TEMP 0.07f
#define INV_T (1.0f / 0.07f)
#define SHIFT (1.0f / 0.07f)  // logits max = sim_max/T = 1/T (lse shift-invariant)

typedef __attribute__((ext_vector_type(16))) __bf16 v16bf;
typedef __attribute__((ext_vector_type(8)))  __bf16 v8bf;
typedef __attribute__((ext_vector_type(4)))  __bf16 v4bf;
typedef __attribute__((ext_vector_type(8)))  float  v8f;

// ---------------- workspace layout (bytes) ----------------
#define WS_PROTO   0                                   // f32 [KK][DD]     102400
#define WS_ZN      ((WS_PROTO + KK*DD*4 + 255) & ~255) // bf16 [NN][DD]   3211264
#define WS_RPOINT  ((WS_ZN + NN*DD*2 + 255) & ~255)    // f32 [NN]
#define WS_DENOM   ((WS_RPOINT + NN*4 + 255) & ~255)   // f32 [NN]
#define WS_POSSUM  ((WS_DENOM + NN*4 + 255) & ~255)    // f32 [NN]
#define WS_POSCNT  ((WS_POSSUM + NN*4 + 255) & ~255)   // f32 [NN]
#define WS_BLOSS   ((WS_POSCNT + NN*4 + 255) & ~255)   // f32 [BB]

// ============ K1: normalize prototypes ============
__global__ void norm_proto_kernel(const float* __restrict__ proto,
                                  float* __restrict__ protoN) {
    int k = blockIdx.x * blockDim.x + threadIdx.x;
    if (k >= KK) return;
    const float4* row = (const float4*)(proto + (size_t)k * DD);
    float nrm = 0.f;
    #pragma unroll 4
    for (int d = 0; d < DD / 4; ++d) {
        float4 v = row[d];
        nrm += v.x * v.x + v.y * v.y + v.z * v.z + v.w * v.w;
    }
    float rinv = rsqrtf(nrm);
    float4* out = (float4*)(protoN + (size_t)k * DD);
    #pragma unroll 4
    for (int d = 0; d < DD / 4; ++d) {
        float4 v = row[d];
        v.x *= rinv; v.y *= rinv; v.z *= rinv; v.w *= rinv;
        out[d] = v;
    }
}

// ============ K2: normalize z -> bf16, compute Rpoint ============
__global__ void norm_z_rpoint_kernel(const float* __restrict__ z,
                                     const float* __restrict__ protoN,
                                     __bf16* __restrict__ zn,
                                     float* __restrict__ rpoint) {
    int i = blockIdx.x * blockDim.x + threadIdx.x;
    if (i >= NN) return;
    const float4* row = (const float4*)(z + (size_t)i * DD);
    float nrm = 0.f;
    #pragma unroll 4
    for (int d = 0; d < DD / 4; ++d) {
        float4 v = row[d];
        nrm += v.x * v.x + v.y * v.y + v.z * v.z + v.w * v.w;
    }
    float rinv = rsqrtf(nrm);
    // write normalized bf16 row (for the WMMA sim GEMM)
    __bf16* outr = zn + (size_t)i * DD;
    #pragma unroll 4
    for (int d = 0; d < DD / 4; ++d) {
        float4 v = row[d];
        v4bf b;
        b[0] = (__bf16)(v.x * rinv); b[1] = (__bf16)(v.y * rinv);
        b[2] = (__bf16)(v.z * rinv); b[3] = (__bf16)(v.w * rinv);
        *(v4bf*)(outr + d * 4) = b;
    }
    // Rpoint[i] = max_k  (z_i . p_k) * rinv     (p already normalized, f32)
    float best = -3.4e38f;
    for (int k = 0; k < KK; ++k) {
        const float4* pr = (const float4*)(protoN + (size_t)k * DD);
        float acc = 0.f;
        #pragma unroll 4
        for (int d = 0; d < DD / 4; ++d) {
            float4 a = row[d];
            float4 b = pr[d];
            acc += a.x * b.x + a.y * b.y + a.z * b.z + a.w * b.w;
        }
        acc *= rinv;
        best = fmaxf(best, acc);
    }
    rpoint[i] = best;
}

// ============ K3: fused sim GEMM (bf16 WMMA) + masking + exp + row sums ======
// grid = NTILES blocks of 256 threads (8 waves). Block owns row tile `it`;
// wave w streams column tiles jt = w, w+8, ... (392 = 8*49, wave-uniform, so
// EXEC stays all-ones around every WMMA). Two independent accumulator chains
// (even/odd k-steps) double WMMA ILP; prefetch hides the B-stream L2 latency.
__global__ __launch_bounds__(256)
void sim_fused_kernel(const __bf16* __restrict__ zn,
                      const float* __restrict__ rpoint,
                      const unsigned char* __restrict__ negl,   // [SS][NN] bool
                      float* __restrict__ denom,
                      float* __restrict__ possum,
                      float* __restrict__ poscnt) {
    __shared__ float s_d[16], s_p[16], s_c[16];
    const int t    = threadIdx.x;
    const int lane = t & 31;
    const int wave = t >> 5;
    const int half = lane >> 4;        // 0: lanes 0-15, 1: lanes 16-31
    const int n16  = lane & 15;
    if (t < 16) { s_d[t] = 0.f; s_p[t] = 0.f; s_c[t] = 0.f; }
    __syncthreads();

    const int it = blockIdx.x;
    const int rowbase = it * 16;

    // ---- preload A fragments for the whole row tile (K = 256 -> 8 frags) ----
    // A layout (16x32 bf16): lane holds row M = lane%16;
    //   lanes 0-15 : K = k0+{0..7}  and k0+{16..23}
    //   lanes 16-31: K = k0+{8..15} and k0+{24..31}
    const __bf16* zr = zn + (size_t)(rowbase + n16) * DD;
    v16bf A[KSTEPS];
    #pragma unroll
    for (int kb = 0; kb < KSTEPS; ++kb) {
        const int k0 = kb * 32 + half * 8;
        v8bf lo = *(const v8bf*)(zr + k0);
        v8bf hi = *(const v8bf*)(zr + k0 + 16);
        A[kb] = __builtin_shufflevector(lo, hi, 0, 1, 2, 3, 4, 5, 6, 7,
                                                8, 9, 10, 11, 12, 13, 14, 15);
    }
    // per-lane row metadata (C layout: VGPR r -> row r + half*8)
    float rp[8]; int rmod[8];
    #pragma unroll
    for (int r = 0; r < 8; ++r) {
        int grow = rowbase + r + half * 8;
        rp[r]   = rpoint[grow];
        rmod[r] = grow % SS;            // column excluded for this row
    }

    float dsum[8], psum[8], pcnt[8];
    #pragma unroll
    for (int r = 0; r < 8; ++r) { dsum[r] = 0.f; psum[r] = 0.f; pcnt[r] = 0.f; }

    for (int jt = wave; jt < NTILES; jt += 8) {
        const int col = jt * 16 + n16;          // my column (same for all 8 rows)
        const float rpj = rpoint[col];
        const __bf16* zc = zn + (size_t)col * DD;
        // prefetch next column tile this wave will touch (col + 16*8 rows on)
        __builtin_prefetch(zc + (size_t)128 * DD, 0, 1);
        // Two independent f32 accumulator chains: even k-steps -> c0,
        // odd k-steps -> c1. Breaks the serial D->C WMMA dependency.
        v8f c0 = {}, c1 = {};
        // B layout (32x16 bf16): lane holds column N = lane%16;
        //   lanes 0-15: K = k0+{0..7} and k0+{16..23}, lanes 16-31: K = k0+{8..15} and k0+{24..31}
        #pragma unroll
        for (int kb = 0; kb < KSTEPS; kb += 2) {
            const int k0a = kb * 32 + half * 8;
            const int k0b = (kb + 1) * 32 + half * 8;
            v8bf alo = *(const v8bf*)(zc + k0a);
            v8bf ahi = *(const v8bf*)(zc + k0a + 16);
            v8bf blo = *(const v8bf*)(zc + k0b);
            v8bf bhi = *(const v8bf*)(zc + k0b + 16);
            v16bf Ba = __builtin_shufflevector(alo, ahi, 0, 1, 2, 3, 4, 5, 6, 7,
                                                         8, 9, 10, 11, 12, 13, 14, 15);
            v16bf Bb = __builtin_shufflevector(blo, bhi, 0, 1, 2, 3, 4, 5, 6, 7,
                                                         8, 9, 10, 11, 12, 13, 14, 15);
            c0 = __builtin_amdgcn_wmma_f32_16x16x32_bf16(
                     false, A[kb],     false, Ba, (short)0, c0, false, false);
            c1 = __builtin_amdgcn_wmma_f32_16x16x32_bf16(
                     false, A[kb + 1], false, Bb, (short)0, c1, false, false);
        }
        // ---- fused elementwise: masks + exp + row accumulation ----
        #pragma unroll
        for (int r = 0; r < 8; ++r) {
            const float s  = c0[r] + c1[r];
            const bool  m1 = (rp[r] < s) | (rpj < s);           // mask_one
            const bool  ng = negl[(size_t)rmod[r] * NN + col] != 0;
            const bool  nd = (col != rmod[r]);                  // diag exclusion
            const float lg = s * INV_T - SHIFT;                 // shifted logit
            const float e  = __expf(lg);
            dsum[r] += (nd & (m1 | ng)) ? e : 0.f;
            psum[r] += (nd & m1) ? lg : 0.f;
            pcnt[r] += (nd & m1) ? 1.f : 0.f;
        }
    }
    // combine: 16 columns of a row live in one lane-half -> LDS f32 atomics
    #pragma unroll
    for (int r = 0; r < 8; ++r) {
        const int rl = r + half * 8;
        atomicAdd(&s_d[rl], dsum[r]);
        atomicAdd(&s_p[rl], psum[r]);
        atomicAdd(&s_c[rl], pcnt[r]);
    }
    __syncthreads();
    if (t < 16) {
        denom[rowbase + t]  = s_d[t];
        possum[rowbase + t] = s_p[t];
        poscnt[rowbase + t] = s_c[t];
    }
}

// ============ K4: per-block  -num/den  ============
__global__ void block_loss_kernel(const float* __restrict__ denom,
                                  const float* __restrict__ possum,
                                  const float* __restrict__ poscnt,
                                  float* __restrict__ bloss) {
    __shared__ float sn[256], sd[256];
    const int b = blockIdx.x, t = threadIdx.x;
    float n = 0.f, d = 0.f;
    if (t < SS) {
        const int row = b * SS + t;
        const float dn = denom[row];
        n = possum[row] - poscnt[row] * __logf(dn);
        d = poscnt[row];
    }
    sn[t] = n; sd[t] = d;
    __syncthreads();
    for (int s = 128; s > 0; s >>= 1) {
        if (t < s) { sn[t] += sn[t + s]; sd[t] += sd[t + s]; }
        __syncthreads();
    }
    if (t == 0) bloss[b] = -sn[0] / sd[0];   // TEMP/BASE_TEMP == 1
}

// ============ K5: mean over blocks -> scalar ============
__global__ void final_mean_kernel(const float* __restrict__ bloss,
                                  float* __restrict__ out) {
    if (threadIdx.x == 0 && blockIdx.x == 0) {
        float acc = 0.f;
        #pragma unroll
        for (int b = 0; b < BB; ++b) acc += bloss[b];
        out[0] = acc / (float)BB;
    }
}

extern "C" void kernel_launch(void* const* d_in, const int* in_sizes, int n_in,
                              void* d_out, int out_size, void* d_ws, size_t ws_size,
                              hipStream_t stream) {
    const float*         z     = (const float*)d_in[0];         // [B,S,D] f32
    const float*         proto = (const float*)d_in[1];         // [K,D]   f32
    const unsigned char* negl  = (const unsigned char*)d_in[2]; // [S,N]   bool

    char* ws = (char*)d_ws;
    float*  protoN = (float*)(ws + WS_PROTO);
    __bf16* zn     = (__bf16*)(ws + WS_ZN);
    float*  rpoint = (float*)(ws + WS_RPOINT);
    float*  denom  = (float*)(ws + WS_DENOM);
    float*  possum = (float*)(ws + WS_POSSUM);
    float*  poscnt = (float*)(ws + WS_POSCNT);
    float*  bloss  = (float*)(ws + WS_BLOSS);
    float*  out    = (float*)d_out;

    norm_proto_kernel<<<(KK + 127) / 128, 128, 0, stream>>>(proto, protoN);
    norm_z_rpoint_kernel<<<NN / 128, 128, 0, stream>>>(z, protoN, zn, rpoint);
    sim_fused_kernel<<<NTILES, 256, 0, stream>>>(zn, rpoint, negl,
                                                 denom, possum, poscnt);
    block_loss_kernel<<<BB, 256, 0, stream>>>(denom, possum, poscnt, bloss);
    final_mean_kernel<<<1, 32, 0, stream>>>(bloss, out);
}
